// splineN_85323820303021
// MI455X (gfx1250) — matrix operand order repaired and checked by reading, not compiled, not run on hardware
//
#include <hip/hip_runtime.h>
#include <hip/hip_bf16.h>

typedef __attribute__((ext_vector_type(16))) _Float16 v16h;
typedef __attribute__((ext_vector_type(8)))  float    v8f;

// ---------------------------------------------------------------------------
// Helpers
// ---------------------------------------------------------------------------
__device__ __forceinline__ float wcat48(const float* __restrict__ W1,
                                        const float* __restrict__ root1,
                                        int f, int c) {
  // Concatenated weight [64 x 48]: cols 0..15 = W1[0], 16..31 = W1[1], 32..47 = root1
  // W1 layout: [2,64,16], root1: [64,16]
  return (c < 32) ? W1[(c >> 4) * 1024 + f * 16 + (c & 15)]
                  : root1[f * 16 + (c - 32)];
}

__device__ __forceinline__ float wcat96(const float* __restrict__ W2,
                                        const float* __restrict__ root2,
                                        int f, int c) {
  // Concatenated weight [16 x 96]: cols 0..31 = W2[0], 32..63 = W2[1], 64..95 = root2
  // W2 layout: [2,16,32], root2: [16,32]
  return (c < 64) ? W2[(c >> 5) * 512 + f * 32 + (c & 31)]
                  : root2[f * 32 + (c - 64)];
}

__device__ __forceinline__ void atomAddF(float* p, float v) {
  unsafeAtomicAdd(p, v);   // native global_atomic_add_f32
}

// ---------------------------------------------------------------------------
// Layer 1 node transform: g1[n, 0:48] = x[n, 0:64] @ [W1_k0 | W1_k1 | root1]
// One wave handles a 16-node tile: 2 k-steps x 3 column tiles = 6 WMMAs.
// ---------------------------------------------------------------------------
__global__ void __launch_bounds__(128)
transform1_kernel(const float* __restrict__ x, const float* __restrict__ W1,
                  const float* __restrict__ root1, float* __restrict__ g1, int N) {
  const int lane = threadIdx.x & 31;
  const int wave = threadIdx.x >> 5;
  const int tile = blockIdx.x * 4 + wave;
  const int row0 = tile * 16;
  if (row0 >= N) return;                 // wave-uniform exit

  const int m  = lane & 15;              // row within tile (A), col (B, C/D)
  const int hi = lane >> 4;              // half-wave selector
  int mrow = row0 + m;
  if (mrow >= N) mrow = N - 1;           // clamp (duplicate read, store guarded)

  const v8f z8 = {0.f, 0.f, 0.f, 0.f, 0.f, 0.f, 0.f, 0.f};
  v8f acc[3] = {z8, z8, z8};

  #pragma unroll
  for (int ks = 0; ks < 2; ++ks) {
    // A fragment: A[m][k], lane m holds K in {0..7,16..23} (+8 for hi lanes)
    v16h a;
    const float* xr = x + (size_t)mrow * 64 + ks * 32 + hi * 8;
    #pragma unroll
    for (int h = 0; h < 8; ++h) {
      a[h]     = (_Float16)xr[h];
      a[h + 8] = (_Float16)xr[h + 16];
    }
    #pragma unroll
    for (int ct = 0; ct < 3; ++ct) {
      // B fragment: B[k][n], lane n = col; lane holds K in {0..7,16..23} (+8 for hi lanes)
      v16h b;
      #pragma unroll
      for (int h = 0; h < 16; ++h) {
        b[h] = (_Float16)wcat48(W1, root1, ks * 32 + (h & 7) + 8 * hi + (h >> 3) * 16, ct * 16 + m);
      }
      acc[ct] = __builtin_amdgcn_wmma_f32_16x16x32_f16(
          false, a, false, b, (short)0, acc[ct], false, false);
    }
  }

  // C/D store: lane = col n, VGPR v = row (hi lanes: row+8)
  #pragma unroll
  for (int ct = 0; ct < 3; ++ct) {
    #pragma unroll
    for (int v = 0; v < 8; ++v) {
      int r = row0 + v + hi * 8;
      if (r < N) g1[(size_t)r * 48 + ct * 16 + m] = acc[ct][v];
    }
  }
}

// ---------------------------------------------------------------------------
// Layer 2 node transform: g2[n, 0:96] = h16[n, 0:16] @ [W2_k0 | W2_k1 | root2]
// K=16 zero-padded to 32; 6 column tiles = 6 WMMAs per wave.
// ---------------------------------------------------------------------------
__global__ void __launch_bounds__(128)
transform2_kernel(const _Float16* __restrict__ h16, const float* __restrict__ W2,
                  const float* __restrict__ root2, float* __restrict__ g2, int N) {
  const int lane = threadIdx.x & 31;
  const int wave = threadIdx.x >> 5;
  const int tile = blockIdx.x * 4 + wave;
  const int row0 = tile * 16;
  if (row0 >= N) return;

  const int m  = lane & 15;
  const int hi = lane >> 4;
  int mrow = row0 + m;
  if (mrow >= N) mrow = N - 1;

  // A fragment: valid K = (h<8 ? h : 16+h-8) + 8*hi -> only h<8 carries K<16
  v16h a;
  const _Float16* hr = h16 + (size_t)mrow * 16 + hi * 8;
  #pragma unroll
  for (int h = 0; h < 8; ++h) {
    a[h]     = hr[h];
    a[h + 8] = (_Float16)0.f;            // K=16..31 padding
  }

  const v8f z8 = {0.f, 0.f, 0.f, 0.f, 0.f, 0.f, 0.f, 0.f};
  #pragma unroll
  for (int ct = 0; ct < 6; ++ct) {
    // B fragment: K = (h<8 ? h : 16+h-8) + 8*hi -> only h<8 valid, h>=8 zero pad
    v16h b;
    #pragma unroll
    for (int h = 0; h < 16; ++h) {
      b[h] = (h < 8) ? (_Float16)wcat96(W2, root2, h + 8 * hi, ct * 16 + m)
                     : (_Float16)0.f;
    }
    v8f c = __builtin_amdgcn_wmma_f32_16x16x32_f16(
        false, a, false, b, (short)0, z8, false, false);
    #pragma unroll
    for (int v = 0; v < 8; ++v) {
      int r = row0 + v + hi * 8;
      if (r < N) g2[(size_t)r * 96 + ct * 16 + m] = c[v];
    }
  }
}

// ---------------------------------------------------------------------------
// Edge scatter, layer 1: acc1[dst] += (1-u)*g1[src,0:16] + u*g1[src,16:32]
// ---------------------------------------------------------------------------
__global__ void __launch_bounds__(256)
scatter1_kernel(const long long* __restrict__ ei, const float* __restrict__ ea,
                const float* __restrict__ g1, float* __restrict__ acc1,
                float* __restrict__ deg, int E) {
  int e = blockIdx.x * 256 + threadIdx.x;
  if (e >= E) return;
  long long s = ei[e];
  long long d = ei[(size_t)E + e];
  float u  = ea[e];
  float w0 = 1.0f - u;
  const float4* gs = (const float4*)(g1 + (size_t)s * 48);  // 192B rows, 16B aligned
  float* ad = acc1 + (size_t)d * 16;
  #pragma unroll
  for (int q = 0; q < 4; ++q) {
    float4 a = gs[q];        // k=0 block
    float4 b = gs[q + 4];    // k=1 block
    atomAddF(ad + 4 * q + 0, w0 * a.x + u * b.x);
    atomAddF(ad + 4 * q + 1, w0 * a.y + u * b.y);
    atomAddF(ad + 4 * q + 2, w0 * a.z + u * b.z);
    atomAddF(ad + 4 * q + 3, w0 * a.w + u * b.w);
  }
  atomAddF(deg + d, 1.0f);
}

// ---------------------------------------------------------------------------
// Edge scatter, layer 2: acc2[dst] += (1-u)*g2[src,0:32] + u*g2[src,32:64]
// ---------------------------------------------------------------------------
__global__ void __launch_bounds__(256)
scatter2_kernel(const long long* __restrict__ ei, const float* __restrict__ ea,
                const float* __restrict__ g2, float* __restrict__ acc2, int E) {
  int e = blockIdx.x * 256 + threadIdx.x;
  if (e >= E) return;
  long long s = ei[e];
  long long d = ei[(size_t)E + e];
  float u  = ea[e];
  float w0 = 1.0f - u;
  const float4* gs = (const float4*)(g2 + (size_t)s * 96);  // 384B rows, 16B aligned
  float* ad = acc2 + (size_t)d * 32;
  #pragma unroll
  for (int q = 0; q < 8; ++q) {
    float4 a = gs[q];        // k=0 block (cols 0..31)
    float4 b = gs[q + 8];    // k=1 block (cols 32..63)
    atomAddF(ad + 4 * q + 0, w0 * a.x + u * b.x);
    atomAddF(ad + 4 * q + 1, w0 * a.y + u * b.y);
    atomAddF(ad + 4 * q + 2, w0 * a.z + u * b.z);
    atomAddF(ad + 4 * q + 3, w0 * a.w + u * b.w);
  }
}

// ---------------------------------------------------------------------------
// Finalize layer 1: h = elu(acc1/deg + root-part + bias1), stored f16
// ---------------------------------------------------------------------------
__global__ void __launch_bounds__(256)
finalize1_kernel(const float* __restrict__ acc1, const float* __restrict__ deg,
                 const float* __restrict__ g1, const float* __restrict__ bias1,
                 _Float16* __restrict__ h16, int N) {
  int i = blockIdx.x * 256 + threadIdx.x;
  if (i >= N * 16) return;
  int n = i >> 4, c = i & 15;
  float invd = 1.0f / fmaxf(deg[n], 1.0f);
  float v = acc1[i] * invd + g1[(size_t)n * 48 + 32 + c] + bias1[c];
  v = (v > 0.0f) ? v : (expf(v) - 1.0f);  // ELU
  h16[i] = (_Float16)v;
}

// ---------------------------------------------------------------------------
// Finalize layer 2 + log_softmax over 32 classes
// ---------------------------------------------------------------------------
__global__ void __launch_bounds__(256)
finalize2_kernel(const float* __restrict__ acc2, const float* __restrict__ deg,
                 const float* __restrict__ g2, const float* __restrict__ bias2,
                 float* __restrict__ out, int N) {
  int n = blockIdx.x * 256 + threadIdx.x;
  if (n >= N) return;
  float invd = 1.0f / fmaxf(deg[n], 1.0f);
  float v[32];
  float mx = -3.402823466e38f;
  #pragma unroll
  for (int c = 0; c < 32; ++c) {
    float t = acc2[(size_t)n * 32 + c] * invd + g2[(size_t)n * 96 + 64 + c] + bias2[c];
    v[c] = t;
    mx = fmaxf(mx, t);
  }
  float s = 0.0f;
  #pragma unroll
  for (int c = 0; c < 32; ++c) s += expf(v[c] - mx);
  float l = mx + logf(s);
  #pragma unroll
  for (int c = 0; c < 32; ++c) out[(size_t)n * 32 + c] = v[c] - l;
}

// ---------------------------------------------------------------------------
// Host-side launch
// ---------------------------------------------------------------------------
extern "C" void kernel_launch(void* const* d_in, const int* in_sizes, int n_in,
                              void* d_out, int out_size, void* d_ws, size_t ws_size,
                              hipStream_t stream) {
  const float*     x     = (const float*)d_in[0];
  const long long* ei    = (const long long*)d_in[1];   // int64 edge_index [2,E]
  const float*     ea    = (const float*)d_in[3];       // edge_attr [E,1]
  const float*     W1    = (const float*)d_in[4];       // [2,64,16]
  const float*     root1 = (const float*)d_in[5];       // [64,16]
  const float*     bias1 = (const float*)d_in[6];       // [16]
  const float*     W2    = (const float*)d_in[7];       // [2,16,32]
  const float*     root2 = (const float*)d_in[8];       // [16,32]
  const float*     bias2 = (const float*)d_in[9];       // [32]
  float* out = (float*)d_out;

  const int N = in_sizes[0] / 64;
  const int E = in_sizes[1] / 2;

  // Workspace layout: [acc1 | deg | acc2] (zeroed) | g1 | g2 | h16
  char* ws = (char*)d_ws;
  size_t off = 0;
  float* acc1 = (float*)(ws + off); off += (size_t)N * 16 * sizeof(float);
  float* deg  = (float*)(ws + off); off += (size_t)N * sizeof(float);
  float* acc2 = (float*)(ws + off); off += (size_t)N * 32 * sizeof(float);
  const size_t zero_bytes = off;
  float* g1   = (float*)(ws + off); off += (size_t)N * 48 * sizeof(float);
  float* g2   = (float*)(ws + off); off += (size_t)N * 96 * sizeof(float);
  _Float16* h16 = (_Float16*)(ws + off);

  hipMemsetAsync(d_ws, 0, zero_bytes, stream);

  const int tiles   = (N + 15) / 16;
  const int tblocks = (tiles + 3) / 4;       // 4 waves (16-node tiles) per block

  transform1_kernel<<<tblocks, 128, 0, stream>>>(x, W1, root1, g1, N);
  scatter1_kernel<<<(E + 255) / 256, 256, 0, stream>>>(ei, ea, g1, acc1, deg, E);
  finalize1_kernel<<<(N * 16 + 255) / 256, 256, 0, stream>>>(acc1, deg, g1, bias1, h16, N);
  transform2_kernel<<<tblocks, 128, 0, stream>>>(h16, W2, root2, g2, N);
  scatter2_kernel<<<(E + 255) / 256, 256, 0, stream>>>(ei, ea, g2, acc2, E);
  finalize2_kernel<<<(N + 255) / 256, 256, 0, stream>>>(acc2, deg, g2, bias2, out, N);
}
